// BoxAwareMask2FormerHungarianMatcher_64922725646503
// MI455X (gfx1250) — matrix operand order repaired and checked by reading, not compiled, not run on hardware
//
#include <hip/hip_runtime.h>
#include <hip/hip_bf16.h>

typedef __attribute__((ext_vector_type(16))) _Float16 v16h;
typedef __attribute__((ext_vector_type(8)))  _Float16 v8h;
typedef __attribute__((ext_vector_type(8)))  float    v8f;

#define BB 2
#define QQ 200
#define NN 50
#define CC 134
#define HH 256
#define WW 256
#define PP 12544
#define QPAD 208   // 13 * 16
#define NPAD 64    // 4 * 16
#define MT 13
#define NT 4
#define BIGV 1e10f

#define W_CLASS 2.0f
#define W_MASK  5.0f
#define W_DICE  5.0f
#define W_BBOX  5.0f
#define W_GIOU  2.0f

// ---- workspace layout (bytes, 256-aligned) ----
#define O_PRED   ((size_t)0)                                   // f16 [B,QPAD,P]
#define SZ_PRED  ((size_t)BB * QPAD * PP * 2)                  // 10,436,608
#define O_SIG    (O_PRED + SZ_PRED)                            // f16 [B,QPAD,P]
#define O_TGT    (O_SIG + SZ_PRED)                             // f16 [B,NPAD,P]
#define SZ_TGT   ((size_t)BB * NPAD * PP * 2)                  // 3,211,264
#define O_SUMSP  (O_TGT + SZ_TGT)                              // f32 [B,QPAD]
#define O_SUMSG  (O_SUMSP + 1792)                              // f32 [B,QPAD]
#define O_SUMTG  (O_SUMSG + 1792)                              // f32 [B,NPAD]
#define O_BASE   (O_SUMTG + 512)                               // f32 [B,Q,N]

__device__ inline float bilinear_sample(const float* __restrict__ m, float px, float py) {
    float x = px * (float)WW - 0.5f;
    float y = py * (float)HH - 0.5f;
    float x0f = floorf(x), y0f = floorf(y);
    float wx = x - x0f, wy = y - y0f;
    int x0 = (int)x0f, y0 = (int)y0f;
    float acc = 0.f;
    #pragma unroll
    for (int dy = 0; dy < 2; ++dy) {
        #pragma unroll
        for (int dx = 0; dx < 2; ++dx) {
            int ix = x0 + dx, iy = y0 + dy;
            bool valid = (ix >= 0) && (ix < WW) && (iy >= 0) && (iy < HH);
            int ixc = min(max(ix, 0), WW - 1);
            int iyc = min(max(iy, 0), HH - 1);
            float v = valid ? m[iyc * WW + ixc] : 0.f;
            float w = (dx ? wx : (1.f - wx)) * (dy ? wy : (1.f - wy));
            acc += v * w;
        }
    }
    return acc;
}

// one block per (b, n_padded): sample target masks at points -> f16, plus row sum
__global__ void sample_tgt_kernel(const float* __restrict__ masks,   // [B,N,H,W]
                                  const float* __restrict__ pts,     // [B,P,2]
                                  _Float16* __restrict__ tgt,        // [B,NPAD,P]
                                  float* __restrict__ sumTgt) {      // [B,NPAD]
    int bn = blockIdx.x;
    int b = bn / NPAD, n = bn % NPAD;
    _Float16* out = tgt + (size_t)bn * PP;
    if (n >= NN) {
        for (int p = threadIdx.x; p < PP; p += blockDim.x) out[p] = (_Float16)0.f;
        if (threadIdx.x == 0) sumTgt[bn] = 0.f;
        return;
    }
    const float* m  = masks + (size_t)(b * NN + n) * HH * WW;
    const float* pc = pts + (size_t)b * PP * 2;
    float s = 0.f;
    for (int p = threadIdx.x; p < PP; p += blockDim.x) {
        float v = bilinear_sample(m, pc[2 * p], pc[2 * p + 1]);
        out[p] = (_Float16)v;
        s += v;
    }
    __shared__ float red[256];
    red[threadIdx.x] = s;
    __syncthreads();
    for (int off = 128; off > 0; off >>= 1) {
        if (threadIdx.x < off) red[threadIdx.x] += red[threadIdx.x + off];
        __syncthreads();
    }
    if (threadIdx.x == 0) sumTgt[bn] = red[0];
}

// one block per (b, q_padded): sample pred logits -> f16 logits + f16 sigmoid,
// plus rowsum(softplus) and rowsum(sigmoid)
__global__ void sample_pred_kernel(const float* __restrict__ masks,   // [B,Q,H,W]
                                   const float* __restrict__ pts,     // [B,P,2]
                                   _Float16* __restrict__ pred,       // [B,QPAD,P]
                                   _Float16* __restrict__ sig,        // [B,QPAD,P]
                                   float* __restrict__ sumSP,         // [B,QPAD]
                                   float* __restrict__ sumSG) {       // [B,QPAD]
    int bq = blockIdx.x;
    int b = bq / QPAD, q = bq % QPAD;
    _Float16* op = pred + (size_t)bq * PP;
    _Float16* os = sig  + (size_t)bq * PP;
    if (q >= QQ) {
        for (int p = threadIdx.x; p < PP; p += blockDim.x) { op[p] = (_Float16)0.f; os[p] = (_Float16)0.f; }
        if (threadIdx.x == 0) { sumSP[bq] = 0.f; sumSG[bq] = 0.f; }
        return;
    }
    const float* m  = masks + (size_t)(b * QQ + q) * HH * WW;
    const float* pc = pts + (size_t)b * PP * 2;
    float ssp = 0.f, ssg = 0.f;
    for (int p = threadIdx.x; p < PP; p += blockDim.x) {
        float v = bilinear_sample(m, pc[2 * p], pc[2 * p + 1]);
        float sg = 1.f / (1.f + expf(-v));
        float sp = fmaxf(v, 0.f) + log1pf(expf(-fabsf(v)));   // stable softplus
        op[p] = (_Float16)v;
        os[p] = (_Float16)sg;
        ssp += sp;
        ssg += sg;
    }
    __shared__ float r1[256], r2[256];
    r1[threadIdx.x] = ssp; r2[threadIdx.x] = ssg;
    __syncthreads();
    for (int off = 128; off > 0; off >>= 1) {
        if (threadIdx.x < off) { r1[threadIdx.x] += r1[threadIdx.x + off]; r2[threadIdx.x] += r2[threadIdx.x + off]; }
        __syncthreads();
    }
    if (threadIdx.x == 0) { sumSP[bq] = r1[0]; sumSG[bq] = r2[0]; }
}

// per (b,q,n): class softmax cost + bbox L1 + GIoU -> base cost matrix
__global__ void base_cost_kernel(const float* __restrict__ cls,      // [B,Q,C]
                                 const int*   __restrict__ labels,   // [B,N]
                                 const float* __restrict__ pbox,     // [B,Q,4]
                                 const float* __restrict__ tbox,     // [B,N,4]
                                 float* __restrict__ base) {         // [B,Q,N]
    int idx = blockIdx.x * blockDim.x + threadIdx.x;
    if (idx >= BB * QQ * NN) return;
    int n = idx % NN;
    int q = (idx / NN) % QQ;
    int b = idx / (NN * QQ);

    // class softmax prob at target label
    const float* cl = cls + (size_t)(b * QQ + q) * CC;
    float mx = -3.402823466e38f;
    for (int c = 0; c < CC; ++c) mx = fmaxf(mx, cl[c]);
    float den = 0.f;
    for (int c = 0; c < CC; ++c) den += expf(cl[c] - mx);
    int t = labels[b * NN + n];
    float prob = expf(cl[t] - mx) / den;
    float cost = W_CLASS * (-prob);

    const float* pb = pbox + (size_t)(b * QQ + q) * 4;
    const float* tb = tbox + (size_t)(b * NN + n) * 4;
    float l1 = 0.f;
    #pragma unroll
    for (int i = 0; i < 4; ++i) l1 += fabsf(pb[i] - tb[i]);
    cost += W_BBOX * l1;

    // cxcywh -> xyxy (ordered, clipped to [0,1])
    float px0 = pb[0] - 0.5f * pb[2], px1 = pb[0] + 0.5f * pb[2];
    float py0 = pb[1] - 0.5f * pb[3], py1 = pb[1] + 0.5f * pb[3];
    float ax0 = fminf(fminf(px0, px1), 1.f); ax0 = fmaxf(ax0, 0.f);
    float ax1 = fmaxf(px0, px1); ax1 = fminf(fmaxf(ax1, 0.f), 1.f);
    float ay0 = fminf(fminf(py0, py1), 1.f); ay0 = fmaxf(ay0, 0.f);
    float ay1 = fmaxf(py0, py1); ay1 = fminf(fmaxf(ay1, 0.f), 1.f);

    float tx0 = tb[0] - 0.5f * tb[2], tx1 = tb[0] + 0.5f * tb[2];
    float ty0 = tb[1] - 0.5f * tb[3], ty1 = tb[1] + 0.5f * tb[3];
    float bx0 = fminf(fminf(tx0, tx1), 1.f); bx0 = fmaxf(bx0, 0.f);
    float bx1 = fmaxf(tx0, tx1); bx1 = fminf(fmaxf(bx1, 0.f), 1.f);
    float by0 = fminf(fminf(ty0, ty1), 1.f); by0 = fmaxf(by0, 0.f);
    float by1 = fmaxf(ty0, ty1); by1 = fminf(fmaxf(by1, 0.f), 1.f);

    float ap = (ax1 - ax0) * (ay1 - ay0);
    float at = (bx1 - bx0) * (by1 - by0);
    float iw = fmaxf(fminf(ax1, bx1) - fmaxf(ax0, bx0), 0.f);
    float ih = fmaxf(fminf(ay1, by1) - fmaxf(ay0, by0), 0.f);
    float inter = iw * ih;
    float uni = ap + at - inter;
    float iou = inter / uni;                               // NaN possible (handled later)
    float cw = fmaxf(fmaxf(ax1, bx1) - fminf(ax0, bx0), 0.f);
    float ch = fmaxf(fmaxf(ay1, by1) - fminf(ay0, by0), 0.f);
    float ac = cw * ch;
    float giou = iou - (ac - uni) / ac;
    cost += W_GIOU * (-giou);

    base[idx] = cost;
}

// Dual f16 WMMA GEMM over K=P: S1 = pred @ tgt^T, S2 = sig(pred) @ tgt^T,
// fused epilogue assembles final cost matrix.
__global__ void wmma_cost_kernel(const _Float16* __restrict__ predA,  // [B,QPAD,P]
                                 const _Float16* __restrict__ sigA,   // [B,QPAD,P]
                                 const _Float16* __restrict__ tgtB,   // [B,NPAD,P]  (row n holds B^T row)
                                 const float* __restrict__ sumSP,
                                 const float* __restrict__ sumSG,
                                 const float* __restrict__ sumTG,
                                 const float* __restrict__ base,
                                 float* __restrict__ out) {           // [B,Q,N]
    const int tilesPerBatch = MT * NT;                 // 52
    int wave = blockIdx.x * (blockDim.x >> 5) + (threadIdx.x >> 5);
    int lane = threadIdx.x & 31;
    int b  = wave / tilesPerBatch;
    int t  = wave % tilesPerBatch;
    int mt = t / NT, nt = t % NT;

    int mcol = lane & 15;          // A: row within tile; B: column within tile
    int g    = lane >> 4;

    const _Float16* arow1 = predA + (size_t)(b * QPAD + mt * 16 + mcol) * PP + g * 8;
    const _Float16* arow2 = sigA  + (size_t)(b * QPAD + mt * 16 + mcol) * PP + g * 8;
    const _Float16* brow  = tgtB  + (size_t)(b * NPAD + nt * 16 + mcol) * PP + g * 8;

    v8f c1 = {};
    v8f c2 = {};

    #pragma unroll 2
    for (int k0 = 0; k0 < PP; k0 += 32) {
        v8h a1lo = *(const v8h*)(arow1 + k0);
        v8h a1hi = *(const v8h*)(arow1 + k0 + 16);
        v8h a2lo = *(const v8h*)(arow2 + k0);
        v8h a2hi = *(const v8h*)(arow2 + k0 + 16);
        v8h blo  = *(const v8h*)(brow + k0);
        v8h bhi  = *(const v8h*)(brow + k0 + 16);
        v16h a1, a2, bf;
        #pragma unroll
        for (int i = 0; i < 8; ++i) {
            a1[i] = a1lo[i]; a1[i + 8] = a1hi[i];
            a2[i] = a2lo[i]; a2[i + 8] = a2hi[i];
            bf[i] = blo[i];  bf[i + 8] = bhi[i];
        }
        c1 = __builtin_amdgcn_wmma_f32_16x16x32_f16(false, a1, false, bf, (short)0, c1, false, false);
        c2 = __builtin_amdgcn_wmma_f32_16x16x32_f16(false, a2, false, bf, (short)0, c2, false, false);
    }

    // C/D layout: VGPR r, lanes 0-15 -> (M=r, N=lane); lanes 16-31 -> (M=r+8, N=lane-16)
    int nIdx  = nt * 16 + mcol;
    int rbase = g * 8;
    const float invP = 1.f / (float)PP;
    #pragma unroll
    for (int r = 0; r < 8; ++r) {
        int q = mt * 16 + rbase + r;
        if (q < QQ && nIdx < NN) {
            float S1 = c1[r];
            float S2 = c2[r];
            float bce  = (sumSP[b * QPAD + q] - S1) * invP;
            float dice = 1.f - (2.f * S2 + 1.f) /
                               (sumSG[b * QPAD + q] + sumTG[b * NPAD + nIdx] + 1.f);
            float v = base[(size_t)(b * QQ + q) * NN + nIdx] + W_MASK * bce + W_DICE * dice;
            if (v != v) v = 0.f;                       // nan_to_num(nan=0)
            v = fminf(fmaxf(v, -BIGV), BIGV);          // +/-inf and clip -> +/-BIG
            out[(size_t)(b * QQ + q) * NN + nIdx] = v;
        }
    }
}

extern "C" void kernel_launch(void* const* d_in, const int* in_sizes, int n_in,
                              void* d_out, int out_size, void* d_ws, size_t ws_size,
                              hipStream_t stream) {
    const float* mask_q  = (const float*)d_in[0];   // [B,Q,H,W]
    const float* cls_q   = (const float*)d_in[1];   // [B,Q,C]
    const float* mask_t  = (const float*)d_in[2];   // [B,N,H,W]
    const float* box_q   = (const float*)d_in[3];   // [B,Q,4]
    const float* box_t   = (const float*)d_in[4];   // [B,N,4]
    const float* pts     = (const float*)d_in[5];   // [B,P,2]
    const int*   labels  = (const int*)d_in[6];     // [B,N]

    char* ws = (char*)d_ws;
    _Float16* wPred = (_Float16*)(ws + O_PRED);
    _Float16* wSig  = (_Float16*)(ws + O_SIG);
    _Float16* wTgt  = (_Float16*)(ws + O_TGT);
    float* sSP  = (float*)(ws + O_SUMSP);
    float* sSG  = (float*)(ws + O_SUMSG);
    float* sTG  = (float*)(ws + O_SUMTG);
    float* bCst = (float*)(ws + O_BASE);

    sample_tgt_kernel<<<BB * NPAD, 256, 0, stream>>>(mask_t, pts, wTgt, sTG);
    sample_pred_kernel<<<BB * QPAD, 256, 0, stream>>>(mask_q, pts, wPred, wSig, sSP, sSG);
    base_cost_kernel<<<(BB * QQ * NN + 255) / 256, 256, 0, stream>>>(cls_q, labels, box_q, box_t, bCst);

    // 104 tiles total (2 batches * 13 * 4); 4 waves/block * 26 blocks, no partial waves
    wmma_cost_kernel<<<26, 128, 0, stream>>>(wPred, wSig, wTgt, sSP, sSG, sTG, bCst, (float*)d_out);
}
